// BasicBlock_73143293051462
// MI455X (gfx1250) — matrix operand order repaired and checked by reading, not compiled, not run on hardware
//
#include <hip/hip_runtime.h>

// Problem constants
#define BB  16
#define CCH 32
#define HH  256
#define WW  256
#define HP  258   // padded height
#define WPD 258   // padded width

typedef __attribute__((ext_vector_type(8)))  unsigned short ushort8;
typedef __attribute__((ext_vector_type(16))) unsigned short ushort16;
typedef __attribute__((ext_vector_type(16))) __bf16         v16bf;
typedef __attribute__((ext_vector_type(8)))  float          v8f;
typedef __attribute__((ext_vector_type(4)))  float          float4v;

__device__ __forceinline__ unsigned short f2bf(float f) {
  unsigned u = __builtin_bit_cast(unsigned, f);
  u += 0x7FFFu + ((u >> 16) & 1u);   // round to nearest even
  return (unsigned short)(u >> 16);
}

// ---------------------------------------------------------------------------
// Pack weights: w[co][ci][ky][kx] fp32 -> wp[tap][co][ci] bf16 (tap = ky*3+kx)
// ---------------------------------------------------------------------------
__global__ void pack_w_kernel(const float* __restrict__ w1, const float* __restrict__ w2,
                              unsigned short* __restrict__ wp1, unsigned short* __restrict__ wp2) {
  int t = blockIdx.x * blockDim.x + threadIdx.x;
  if (t >= 2 * 9 * 32 * 32) return;
  int sel = t / (9 * 32 * 32);
  int idx = t - sel * (9 * 32 * 32);
  int ci  = idx & 31;
  int co  = (idx >> 5) & 31;
  int tap = idx >> 10;
  const float* w = sel ? w2 : w1;
  unsigned short* wp = sel ? wp2 : wp1;
  wp[idx] = f2bf(w[(co * 32 + ci) * 9 + tap]);
}

// ---------------------------------------------------------------------------
// Pack input: x NCHW fp32 -> xp padded NHWC bf16; zero borders of xp AND hp.
// One thread per padded element; channels fastest -> fully coalesced stores.
// ---------------------------------------------------------------------------
__global__ void pack_x_kernel(const float* __restrict__ x,
                              unsigned short* __restrict__ xp,
                              unsigned short* __restrict__ hp) {
  int t = blockIdx.x * blockDim.x + threadIdx.x;
  int c   = t & 31;
  int pix = t >> 5;
  int xc  = pix % WPD;
  int r   = pix / WPD;
  int yc  = r % HP;
  int b   = r / HP;
  if (b >= BB) return;
  if (xc == 0 || xc == WPD - 1 || yc == 0 || yc == HP - 1) {
    xp[t] = 0;
    hp[t] = 0;   // zero halo of intermediate buffer too
  } else {
    xp[t] = f2bf(x[(((b * CCH + c) * HH) + (yc - 1)) * WW + (xc - 1)]);
  }
}

// ---------------------------------------------------------------------------
// Conv layer 1: xp (padded NHWC bf16) -> hp (padded NHWC bf16)
// per-wave tile: M=16 pixels along W, N=32 output channels, K-loop = 9 taps x 32 ch
// epilogue: relu((acc + bias) * relu(gate)), LDS transpose, b128 stores
// ---------------------------------------------------------------------------
__global__ __launch_bounds__(256) void conv1_kernel(
    const unsigned short* __restrict__ xp, const unsigned short* __restrict__ wp,
    const float* __restrict__ bias, const float* __restrict__ gate,
    unsigned short* __restrict__ hp) {
  __shared__ __align__(32) unsigned short lds[8][16][32];

  const int lane = threadIdx.x & 31;
  const int wib  = threadIdx.x >> 5;
  const int wid  = blockIdx.x * 8 + wib;     // 65536 strips total
  const int m    = lane & 15;                // row (pixel) / column (co) selector
  const int h    = lane >> 4;                // lane half
  const int xt   = wid & 15;
  const int y    = (wid >> 4) & 255;
  const int b    = wid >> 12;
  const int x0   = xt * 16;

  const unsigned short* xbase = xp + (size_t)((b * HP + y) * WPD + x0) * CCH;

  v8f acc0 = {}; v8f acc1 = {};
  #pragma unroll
  for (int tap = 0; tap < 9; ++tap) {
    const int dy = tap / 3, dx = tap % 3;
    // A fragment (16x32 bf16): lane-half h holds K = 16h..16h+15
    const unsigned short* ap = xbase + (size_t)((dy * WPD + dx + m) * CCH) + 16 * h;
    union { ushort8 p[2]; ushort16 w; } ua;
    ua.p[0] = *(const ushort8*)ap;
    ua.p[1] = *(const ushort8*)(ap + 8);
    v16bf a = __builtin_bit_cast(v16bf, ua.w);
    // B fragments (32x16 bf16): lane-half h holds K = 16h..16h+15, column = m (+16)
    const unsigned short* wb0 = wp + (size_t)((tap * 32 + m) * 32) + 16 * h;
    v16bf bf0 = __builtin_bit_cast(v16bf, *(const ushort16*)wb0);
    v16bf bf1 = __builtin_bit_cast(v16bf, *(const ushort16*)(wb0 + 16 * 32));
    acc0 = __builtin_amdgcn_wmma_f32_16x16x32_bf16(false, a, false, bf0, (short)0, acc0, false, false);
    acc1 = __builtin_amdgcn_wmma_f32_16x16x32_bf16(false, a, false, bf1, (short)0, acc1, false, false);
  }

  const float g0  = fmaxf(gate[b * 32 + m], 0.f);
  const float g1  = fmaxf(gate[b * 32 + m + 16], 0.f);
  const float bb0 = bias[m], bb1 = bias[m + 16];
  #pragma unroll
  for (int v = 0; v < 8; ++v) {
    lds[wib][v + 8 * h][m]      = f2bf(fmaxf((acc0[v] + bb0) * g0, 0.f));
    lds[wib][v + 8 * h][m + 16] = f2bf(fmaxf((acc1[v] + bb1) * g1, 0.f));
  }
  __syncthreads();
  // each lane emits half a pixel's channel row: 32B contiguous
  const int p = lane >> 1, q = lane & 1;
  ushort16 row = *(const ushort16*)&lds[wib][p][q * 16];
  unsigned short* dst = hp + (size_t)((b * HP + y + 1) * WPD + (x0 + p + 1)) * CCH + q * 16;
  *(ushort16*)dst = row;
}

// ---------------------------------------------------------------------------
// Conv layer 2: hp (padded NHWC bf16) -> out NCHW fp32, with residual x
// ---------------------------------------------------------------------------
__global__ __launch_bounds__(256) void conv2_kernel(
    const unsigned short* __restrict__ hp, const unsigned short* __restrict__ wp,
    const float* __restrict__ bias, const float* __restrict__ gate,
    const float* __restrict__ xres, float* __restrict__ out) {
  const int lane = threadIdx.x & 31;
  const int wib  = threadIdx.x >> 5;
  const int wid  = blockIdx.x * 8 + wib;
  const int m    = lane & 15;
  const int h    = lane >> 4;
  const int xt   = wid & 15;
  const int y    = (wid >> 4) & 255;
  const int b    = wid >> 12;
  const int x0   = xt * 16;

  const unsigned short* xbase = hp + (size_t)((b * HP + y) * WPD + x0) * CCH;

  v8f acc0 = {}; v8f acc1 = {};
  #pragma unroll
  for (int tap = 0; tap < 9; ++tap) {
    const int dy = tap / 3, dx = tap % 3;
    const unsigned short* ap = xbase + (size_t)((dy * WPD + dx + m) * CCH) + 16 * h;
    union { ushort8 p[2]; ushort16 w; } ua;
    ua.p[0] = *(const ushort8*)ap;
    ua.p[1] = *(const ushort8*)(ap + 8);
    v16bf a = __builtin_bit_cast(v16bf, ua.w);
    const unsigned short* wb0 = wp + (size_t)((tap * 32 + m) * 32) + 16 * h;
    v16bf bf0 = __builtin_bit_cast(v16bf, *(const ushort16*)wb0);
    v16bf bf1 = __builtin_bit_cast(v16bf, *(const ushort16*)(wb0 + 16 * 32));
    acc0 = __builtin_amdgcn_wmma_f32_16x16x32_bf16(false, a, false, bf0, (short)0, acc0, false, false);
    acc1 = __builtin_amdgcn_wmma_f32_16x16x32_bf16(false, a, false, bf1, (short)0, acc1, false, false);
  }

  const float g0  = fmaxf(gate[b * 32 + m], 0.f);
  const float g1  = fmaxf(gate[b * 32 + m + 16], 0.f);
  const float bb0 = bias[m], bb1 = bias[m + 16];

  // NCHW addresses: lane column = output channel; rows = 8 consecutive x positions
  const size_t o0 = (((size_t)(b * CCH + m)) * HH + y) * WW + x0 + 8 * h;
  const size_t o1 = (((size_t)(b * CCH + m + 16)) * HH + y) * WW + x0 + 8 * h;

  union { float4v v[2]; float f[8]; } r0, r1, s0, s1;
  r0.v[0] = ((const float4v*)(xres + o0))[0];
  r0.v[1] = ((const float4v*)(xres + o0))[1];
  r1.v[0] = ((const float4v*)(xres + o1))[0];
  r1.v[1] = ((const float4v*)(xres + o1))[1];
  #pragma unroll
  for (int v = 0; v < 8; ++v) {
    s0.f[v] = fmaxf((acc0[v] + bb0) * g0, 0.f) + r0.f[v];
    s1.f[v] = fmaxf((acc1[v] + bb1) * g1, 0.f) + r1.f[v];
  }
  ((float4v*)(out + o0))[0] = s0.v[0];
  ((float4v*)(out + o0))[1] = s0.v[1];
  ((float4v*)(out + o1))[0] = s1.v[0];
  ((float4v*)(out + o1))[1] = s1.v[1];
}

// ---------------------------------------------------------------------------
extern "C" void kernel_launch(void* const* d_in, const int* in_sizes, int n_in,
                              void* d_out, int out_size, void* d_ws, size_t ws_size,
                              hipStream_t stream) {
  const float* x    = (const float*)d_in[0];
  const float* gate = (const float*)d_in[1];
  const float* w1   = (const float*)d_in[2];
  const float* b1   = (const float*)d_in[3];
  const float* w2   = (const float*)d_in[4];
  const float* b2   = (const float*)d_in[5];
  float* out = (float*)d_out;

  const size_t padElems = (size_t)BB * HP * WPD * CCH;  // 34,080,768 bf16 elems
  unsigned short* xp  = (unsigned short*)d_ws;
  unsigned short* hpb = xp + padElems;
  unsigned short* wp1 = hpb + padElems;
  unsigned short* wp2 = wp1 + 9 * 32 * 32;

  pack_w_kernel<<<(2 * 9 * 32 * 32 + 255) / 256, 256, 0, stream>>>(w1, w2, wp1, wp2);
  pack_x_kernel<<<(int)(padElems / 256), 256, 0, stream>>>(x, xp, hpb);
  conv1_kernel<<<8192, 256, 0, stream>>>(xp, wp1, b1, gate, hpb);
  conv2_kernel<<<8192, 256, 0, stream>>>(hpb, wp2, b2, gate, x, out);
}
